// UAVAttentionNetwork_88441966559608
// MI455X (gfx1250) — compile-verified, matrix operand order checked by reading, not run on hardware
//
#include <hip/hip_runtime.h>
#include <hip/hip_bf16.h>
#include <math.h>

// ---------------------------------------------------------------------------
// CDNA5 (gfx1250) implementation of the UAV GAT network.
// wave32, WMMA 16x16x32 f16 (f32 accumulate) for every dense GEMM.
// B operands are pre-swizzled once into the WMMA B-register layout (f16),
// so the GEMM inner loop does one contiguous 32B load per lane for B.
// The HBM-bound pooling GEMM is K-split 8x for memory-level parallelism.
// ---------------------------------------------------------------------------

typedef __attribute__((ext_vector_type(16))) _Float16 v16h;
typedef __attribute__((ext_vector_type(8)))  float    v8f;

#define HEADS  4
#define HDIM   64
#define KSPLIT 8          // K-parallelism for the 320MB adj stream

// ---- ordered-uint encoding for float atomic max -----------------------------
__device__ __forceinline__ unsigned fenc(float f) {
    unsigned u = __float_as_uint(f);
    return (u & 0x80000000u) ? ~u : (u | 0x80000000u);
}
__device__ __forceinline__ float fdec(unsigned u) {
    return (u & 0x80000000u) ? __uint_as_float(u & 0x7fffffffu) : __uint_as_float(~u);
}
#define ENC_NEG_INF 0x007FFFFFu   // fenc(-inf)

// ---------------------------------------------------------------------------
// WMMA A-operand packing (ISA 7.12.2 layout, wave32).
// ---------------------------------------------------------------------------
__device__ __forceinline__ v16h pack_a_fast(const float* __restrict__ A, int lda,
                                            int row0, int k0, int lane) {
    int m  = row0 + (lane & 15);
    int hi = lane >> 4;
    const float* p = A + (size_t)m * lda + k0 + hi * 8;
    float4 l0 = *(const float4*)(p);
    float4 l1 = *(const float4*)(p + 4);
    float4 h0 = *(const float4*)(p + 16);
    float4 h1 = *(const float4*)(p + 20);
    v16h a;
    a[0]  = (_Float16)l0.x; a[1]  = (_Float16)l0.y; a[2]  = (_Float16)l0.z; a[3]  = (_Float16)l0.w;
    a[4]  = (_Float16)l1.x; a[5]  = (_Float16)l1.y; a[6]  = (_Float16)l1.z; a[7]  = (_Float16)l1.w;
    a[8]  = (_Float16)h0.x; a[9]  = (_Float16)h0.y; a[10] = (_Float16)h0.z; a[11] = (_Float16)h0.w;
    a[12] = (_Float16)h1.x; a[13] = (_Float16)h1.y; a[14] = (_Float16)h1.z; a[15] = (_Float16)h1.w;
    return a;
}

__device__ __forceinline__ v16h pack_a_safe(const float* __restrict__ A, int lda,
                                            int row0, int k0, int M, int K, int lane) {
    int m  = row0 + (lane & 15);
    int hi = lane >> 4;
    bool mok = (m < M);
    int  mc  = mok ? m : 0;
    const float* p = A + (size_t)mc * lda;
    v16h a;
#pragma unroll
    for (int e = 0; e < 8; ++e) {
        int k  = k0 + hi * 8 + e;
        int k2 = k + 16;
        bool ok1 = mok && (k  < K);
        bool ok2 = mok && (k2 < K);
        float v1 = p[ok1 ? k  : 0];   // clamped address: load always legal
        float v2 = p[ok2 ? k2 : 0];
        a[e]     = (_Float16)(ok1 ? v1 : 0.f);
        a[8 + e] = (_Float16)(ok2 ? v2 : 0.f);
    }
    return a;
}

// ---------------------------------------------------------------------------
// B operand pre-swizzle: f32 row-major [K,N] -> f16 tiles in WMMA B-register
// order. Tile (kt,nt) stores, per lane, the 16 f16 values that lane needs:
// one contiguous 32-byte load in the GEMM loop. K/N tails zero-padded (SAFE).
// ---------------------------------------------------------------------------
template <bool SAFE>
__global__ void b_swizzle_kernel(const float* __restrict__ B, _Float16* __restrict__ Bs,
                                 int K, int N) {
    int tilesN = (N + 15) >> 4;
    int tilesK = (K + 31) >> 5;
    int total  = tilesK * tilesN * 32;
    int i = blockIdx.x * blockDim.x + threadIdx.x;
    if (i >= total) return;
    int lane = i & 31;
    int tile = i >> 5;
    int nt = tile % tilesN;
    int kt = tile / tilesN;
    int nn = nt * 16 + (lane & 15);
    int hi = lane >> 4;
    int k0 = kt * 32;
    v16h b;
    if constexpr (!SAFE) {
        const float* p = B + (size_t)(k0 + hi * 8) * N + nn;
#pragma unroll
        for (int e = 0; e < 8; ++e) {
            b[e]     = (_Float16)p[(size_t)e * N];
            b[8 + e] = (_Float16)p[(size_t)(e + 16) * N];
        }
    } else {
        bool nok = (nn < N);
        int  nc  = nok ? nn : 0;
#pragma unroll
        for (int e = 0; e < 8; ++e) {
            int k  = k0 + hi * 8 + e;
            int k2 = k + 16;
            bool ok1 = nok && (k  < K);
            bool ok2 = nok && (k2 < K);
            float v1 = B[(size_t)(ok1 ? k  : 0) * N + nc];
            float v2 = B[(size_t)(ok2 ? k2 : 0) * N + nc];
            b[e]     = (_Float16)(ok1 ? v1 : 0.f);
            b[8 + e] = (_Float16)(ok2 ? v2 : 0.f);
        }
    }
    *(v16h*)(Bs + ((size_t)tile * 32 + lane) * 16) = b;
}

__device__ __forceinline__ v16h pack_b16(const _Float16* __restrict__ Bs, int tilesN,
                                         int kt, int nt, int lane) {
    return *(const v16h*)(Bs + (((size_t)kt * tilesN + nt) * 32 + lane) * 16);
}

// ---------------------------------------------------------------------------
// Fast GEMM: C[M,N] = A[M,K] @ B[K,N] (+bias). One wave per 16x64 strip:
// A tile packed once per 4 WMMAs. Requires M%16==0, K%32==0, N%64==0.
// ---------------------------------------------------------------------------
__global__ void wmma_gemm_fast_kernel(const float* __restrict__ A, const _Float16* __restrict__ Bs,
                                      float* __restrict__ C, const float* __restrict__ bias,
                                      int M, int N, int K) {
    int wid  = (int)((blockIdx.x * (long long)blockDim.x + threadIdx.x) >> 5);
    int lane = threadIdx.x & 31;
    int groupsN = N >> 6;
    int tilesN  = N >> 4;
    int tilesM  = M >> 4;
    if (wid >= tilesM * groupsN) return;          // wave-uniform
    int row0 = (wid / groupsN) * 16;
    int nt0  = (wid % groupsN) * 4;
    v8f c0 = {}, c1 = {}, c2 = {}, c3 = {};
    for (int kt = 0; kt < (K >> 5); ++kt) {
        v16h a  = pack_a_fast(A, K, row0, kt * 32, lane);
        v16h b0 = pack_b16(Bs, tilesN, kt, nt0,     lane);
        v16h b1 = pack_b16(Bs, tilesN, kt, nt0 + 1, lane);
        v16h b2 = pack_b16(Bs, tilesN, kt, nt0 + 2, lane);
        v16h b3 = pack_b16(Bs, tilesN, kt, nt0 + 3, lane);
        c0 = __builtin_amdgcn_wmma_f32_16x16x32_f16(false, a, false, b0, (short)0, c0, false, false);
        c1 = __builtin_amdgcn_wmma_f32_16x16x32_f16(false, a, false, b1, (short)0, c1, false, false);
        c2 = __builtin_amdgcn_wmma_f32_16x16x32_f16(false, a, false, b2, (short)0, c2, false, false);
        c3 = __builtin_amdgcn_wmma_f32_16x16x32_f16(false, a, false, b3, (short)0, c3, false, false);
    }
    int hi = lane >> 4;
    int nb = lane & 15;
    int col0 = nt0 * 16;
#pragma unroll
    for (int r = 0; r < 8; ++r) {
        int m = row0 + r + 8 * hi;
        float* dst = C + (size_t)m * N + col0 + nb;
        float b0v = bias ? bias[col0 + nb]      : 0.f;
        float b1v = bias ? bias[col0 + 16 + nb] : 0.f;
        float b2v = bias ? bias[col0 + 32 + nb] : 0.f;
        float b3v = bias ? bias[col0 + 48 + nb] : 0.f;
        dst[0]  = c0[r] + b0v;
        dst[16] = c1[r] + b1v;
        dst[32] = c2[r] + b2v;
        dst[48] = c3[r] + b3v;
    }
}

// ---- generic (safe) GEMM: one wave per 16x16 tile, any M/N/K ----------------
__global__ void wmma_gemm_safe_kernel(const float* __restrict__ A, const _Float16* __restrict__ Bs,
                                      float* __restrict__ C, const float* __restrict__ bias,
                                      int M, int N, int K) {
    int wid  = (int)((blockIdx.x * (long long)blockDim.x + threadIdx.x) >> 5);
    int lane = threadIdx.x & 31;
    int tilesN = (N + 15) >> 4;
    int tilesM = (M + 15) >> 4;
    if (wid >= tilesM * tilesN) return;          // wave-uniform
    int row0 = (wid / tilesN) * 16;
    int nt   = wid % tilesN;
    int tilesK = (K + 31) >> 5;
    v8f c = {};
    for (int kt = 0; kt < tilesK; ++kt) {
        v16h a = pack_a_safe(A, K, row0, kt * 32, M, K, lane);
        v16h b = pack_b16(Bs, tilesN, kt, nt, lane);
        c = __builtin_amdgcn_wmma_f32_16x16x32_f16(false, a, false, b, (short)0, c, false, false);
    }
    int hi = lane >> 4;
    int n  = nt * 16 + (lane & 15);
#pragma unroll
    for (int r = 0; r < 8; ++r) {
        int m = row0 + r + 8 * hi;
        if (m < M && n < N) {
            float v = c[r];
            if (bias) v += bias[n];
            C[(size_t)m * N + n] = v;
        }
    }
}

// ---- fill -------------------------------------------------------------------
__global__ void fill_u32_kernel(unsigned* p, unsigned v, long long nElems) {
    long long i = blockIdx.x * (long long)blockDim.x + threadIdx.x;
    if (i < nElems) p[i] = v;
}

// ---- per-node attention scores ---------------------------------------------
__global__ void att_kernel(const float* __restrict__ h, const float* __restrict__ att_s,
                           const float* __restrict__ att_d, float* __restrict__ a_s,
                           float* __restrict__ a_d, int n, int heads) {
    int i = blockIdx.x * blockDim.x + threadIdx.x;
    if (i >= n * heads) return;
    int node = i / heads, hd = i % heads;
    const float* hp = h + (size_t)node * heads * HDIM + hd * HDIM;
    float ss = 0.f, sd = 0.f;
#pragma unroll 4
    for (int f = 0; f < HDIM; ++f) {
        ss += hp[f] * att_s[hd * HDIM + f];
        sd += hp[f] * att_d[hd * HDIM + f];
    }
    a_s[i] = ss;
    a_d[i] = sd;
}

// ---- edge pass A: e = leaky_relu(a_s[src]+a_d[dst]); segment max ------------
__global__ void edge_max_kernel(const int* __restrict__ src, const int* __restrict__ dst,
                                const float* __restrict__ a_s, const float* __restrict__ a_d,
                                float* __restrict__ e_buf, unsigned* __restrict__ mEnc,
                                long long ne, int heads) {
    long long i = blockIdx.x * (long long)blockDim.x + threadIdx.x;
    if (i >= ne * heads) return;
    long long eid = i / heads;
    int hd = (int)(i % heads);
    int s = src[eid], d = dst[eid];
    float v = a_s[(size_t)s * heads + hd] + a_d[(size_t)d * heads + hd];
    v = v > 0.f ? v : 0.2f * v;
    e_buf[i] = v;
    atomicMax(&mEnc[(size_t)d * heads + hd], fenc(v));
}

// ---- edge pass B: ex = exp(e - m[dst]); segment sum denom -------------------
__global__ void edge_exp_kernel(const int* __restrict__ dst, float* __restrict__ e_buf,
                                const unsigned* __restrict__ mEnc, float* __restrict__ denom,
                                long long ne, int heads) {
    long long i = blockIdx.x * (long long)blockDim.x + threadIdx.x;
    if (i >= ne * heads) return;
    long long eid = i / heads;
    int hd = (int)(i % heads);
    int d = dst[eid];
    float mm = fdec(mEnc[(size_t)d * heads + hd]);
    float v = __expf(e_buf[i] - mm);
    e_buf[i] = v;
    atomicAdd(&denom[(size_t)d * heads + hd], v);
}

// ---- edge pass C: scatter msg = h[src] * alpha into x[dst] ------------------
__global__ void edge_msg_kernel(const int* __restrict__ src, const int* __restrict__ dst,
                                const float* __restrict__ e_buf, const float* __restrict__ denom,
                                const float* __restrict__ h, float* __restrict__ xout,
                                long long ne, int heads) {
    long long total = ne * heads * HDIM;
    long long i = blockIdx.x * (long long)blockDim.x + threadIdx.x;
    if (i >= total) return;
    int fdimH = heads * HDIM;
    long long eid = i / fdimH;
    int r  = (int)(i % fdimH);
    int hd = r / HDIM;
    int s = src[eid], d = dst[eid];
    float alpha = e_buf[eid * heads + hd] / (denom[(size_t)d * heads + hd] + 1e-16f);
    atomicAdd(&xout[(size_t)d * fdimH + r], h[(size_t)s * fdimH + r] * alpha);
}

// ---- batch-norm stats (biased var), one block per column --------------------
__global__ void bn_stats_kernel(const float* __restrict__ x, float* __restrict__ mean,
                                float* __restrict__ var, int rows, int cols) {
    int col = blockIdx.x;
    float s = 0.f, s2 = 0.f;
    for (int r = threadIdx.x; r < rows; r += blockDim.x) {
        float v = x[(size_t)r * cols + col];
        s += v; s2 += v * v;
    }
    __shared__ float sh[256], sh2[256];
    sh[threadIdx.x] = s; sh2[threadIdx.x] = s2;
    __syncthreads();
    for (int o = 128; o > 0; o >>= 1) {
        if ((int)threadIdx.x < o) {
            sh[threadIdx.x]  += sh[threadIdx.x + o];
            sh2[threadIdx.x] += sh2[threadIdx.x + o];
        }
        __syncthreads();
    }
    if (threadIdx.x == 0) {
        float mu = sh[0] / rows;
        mean[col] = mu;
        var[col]  = sh2[0] / rows - mu * mu;
    }
}

// ---- batch-norm apply + activation (0=elu, 1=relu) --------------------------
__global__ void bn_apply_kernel(const float* __restrict__ x, float* __restrict__ out,
                                const float* __restrict__ mean, const float* __restrict__ var,
                                const float* __restrict__ g, const float* __restrict__ b,
                                int rows, int cols, int ld_out, int act) {
    long long i = blockIdx.x * (long long)blockDim.x + threadIdx.x;
    if (i >= (long long)rows * cols) return;
    int r  = (int)(i / cols);
    int cc = (int)(i % cols);
    float v = (x[i] - mean[cc]) * rsqrtf(var[cc] + 1e-5f) * g[cc] + b[cc];
    v = (act == 0) ? (v > 0.f ? v : __expf(v) - 1.f) : fmaxf(v, 0.f);
    out[(size_t)r * ld_out + cc] = v;
}

// ---------------------------------------------------------------------------
// Fused masked mean-pool, K-split for HBM-stream parallelism:
//   partial(vis @ th) and partial counts -> atomicAdd into tfeat/cntBuf.
// One wave per (16-row tile, K-chunk); adj is still read exactly once total.
// ---------------------------------------------------------------------------
template <bool SAFE>
__global__ void pool_kernel(const float* __restrict__ adj, const _Float16* __restrict__ thS,
                            float* __restrict__ tfeat, float* __restrict__ cntBuf,
                            int n, int t) {
    int wid  = (int)((blockIdx.x * (long long)blockDim.x + threadIdx.x) >> 5);
    int lane = threadIdx.x & 31;
    int tilesM = (n + 15) >> 4;
    if (wid >= tilesM * KSPLIT) return;          // wave-uniform
    int mt    = wid / KSPLIT;
    int chunk = wid % KSPLIT;
    int tilesK = (t + 31) >> 5;
    int ktPer  = (tilesK + KSPLIT - 1) / KSPLIT;
    int ktBeg  = chunk * ktPer;
    int ktEnd  = ktBeg + ktPer < tilesK ? ktBeg + ktPer : tilesK;
    int row0 = mt * 16;
    int m  = row0 + (lane & 15);
    int hi = lane >> 4;
    v8f c0 = {}, c1 = {}, c2 = {}, c3 = {};
    int cnt = 0;
    for (int kt = ktBeg; kt < ktEnd; ++kt) {
        int k0 = kt * 32;
        v16h a;
        if constexpr (!SAFE) {
            const float* p = adj + (size_t)m * t + k0 + hi * 8;
            float4 l0 = *(const float4*)(p);
            float4 l1 = *(const float4*)(p + 4);
            float4 h0 = *(const float4*)(p + 16);
            float4 h1 = *(const float4*)(p + 20);
            float vv[16] = {l0.x, l0.y, l0.z, l0.w, l1.x, l1.y, l1.z, l1.w,
                            h0.x, h0.y, h0.z, h0.w, h1.x, h1.y, h1.z, h1.w};
#pragma unroll
            for (int e = 0; e < 16; ++e) {
                float av = vv[e] > 0.f ? 1.f : 0.f;
                a[e] = (_Float16)av;
                cnt += (vv[e] > 0.f);
            }
            __builtin_prefetch(p + 32, 0, 1);    // stream-ahead on the 320MB operand
        } else {
            bool mok = (m < n);
            const float* p = adj + (size_t)(mok ? m : 0) * t;
#pragma unroll
            for (int e = 0; e < 8; ++e) {
                int k  = k0 + hi * 8 + e;
                int k2 = k + 16;
                bool ok1 = mok && (k  < t);
                bool ok2 = mok && (k2 < t);
                float v1 = p[ok1 ? k  : 0];
                float v2 = p[ok2 ? k2 : 0];
                float a1 = (ok1 && v1 > 0.f) ? 1.f : 0.f;
                float a2 = (ok2 && v2 > 0.f) ? 1.f : 0.f;
                a[e] = (_Float16)a1;  a[8 + e] = (_Float16)a2;
                cnt += (a1 > 0.f) + (a2 > 0.f);
            }
        }
        v16h b0 = pack_b16(thS, 4, kt, 0, lane);
        v16h b1 = pack_b16(thS, 4, kt, 1, lane);
        v16h b2 = pack_b16(thS, 4, kt, 2, lane);
        v16h b3 = pack_b16(thS, 4, kt, 3, lane);
        c0 = __builtin_amdgcn_wmma_f32_16x16x32_f16(false, a, false, b0, (short)0, c0, false, false);
        c1 = __builtin_amdgcn_wmma_f32_16x16x32_f16(false, a, false, b1, (short)0, c1, false, false);
        c2 = __builtin_amdgcn_wmma_f32_16x16x32_f16(false, a, false, b2, (short)0, c2, false, false);
        c3 = __builtin_amdgcn_wmma_f32_16x16x32_f16(false, a, false, b3, (short)0, c3, false, false);
    }
    cnt += __shfl_xor(cnt, 16);                  // both K-halves of each row
    if (lane < 16 && row0 + lane < n)            // per-row partial count
        atomicAdd(&cntBuf[row0 + lane], (float)cnt);
    int nb = lane & 15;
#pragma unroll
    for (int r = 0; r < 8; ++r) {
        int mrow = row0 + r + 8 * hi;
        if (mrow < n) {
            float* d = tfeat + (size_t)mrow * HDIM;
            atomicAdd(d +  0 + nb, c0[r]);
            atomicAdd(d + 16 + nb, c1[r]);
            atomicAdd(d + 32 + nb, c2[r]);
            atomicAdd(d + 48 + nb, c3[r]);
        }
    }
}

// ---- pooling epilogue: combined[:,64:128] = tfeat / max(cnt,1) --------------
__global__ void pool_norm_kernel(const float* __restrict__ tfeat, const float* __restrict__ cntBuf,
                                 float* __restrict__ combined, int n) {
    long long i = blockIdx.x * (long long)blockDim.x + threadIdx.x;
    if (i >= (long long)n * HDIM) return;
    int row = (int)(i / HDIM);
    int col = (int)(i % HDIM);
    combined[(size_t)row * 128 + 64 + col] = tfeat[i] / fmaxf(cntBuf[row], 1.f);
}

// ---- fused output MLP: relu(combined@Wf1+bf1)@Wf2+bf2 -----------------------
template <bool SAFE>
__global__ void mlp_kernel(const float* __restrict__ combined, const _Float16* __restrict__ Wf1S,
                           const float* __restrict__ bf1, const _Float16* __restrict__ Wf2S,
                           const float* __restrict__ bf2, float* __restrict__ out, int n) {
    __shared__ float y1[8][16 * 64];             // per-wave C->A relayout buffer
    int wid   = (int)((blockIdx.x * (long long)blockDim.x + threadIdx.x) >> 5);
    int lane  = threadIdx.x & 31;
    int wslot = threadIdx.x >> 5;
    int row0  = wid * 16;
    int hi    = lane >> 4;
    int tilesM = (n + 15) >> 4;
    bool live = (wid < tilesM);                  // no early return (block barrier below)
    // layer 1: [16,128] @ [128,64] + bf1, relu -> LDS (row-major 16x64)
    for (int nt = 0; nt < 4; ++nt) {
        v8f c = {};
        for (int kt = 0; kt < 4; ++kt) {
            v16h a;
            if constexpr (!SAFE) a = pack_a_fast(combined, 128, live ? row0 : 0, kt * 32, lane);
            else                 a = pack_a_safe(combined, 128, row0, kt * 32, n, 128, lane);
            v16h b = pack_b16(Wf1S, 4, kt, nt, lane);
            c = __builtin_amdgcn_wmma_f32_16x16x32_f16(false, a, false, b, (short)0, c, false, false);
        }
        int col = nt * 16 + (lane & 15);
#pragma unroll
        for (int r = 0; r < 8; ++r)
            y1[wslot][(r + 8 * hi) * 64 + col] = fmaxf(c[r] + bf1[col], 0.f);
    }
    __syncthreads();
    // layer 2: [16,64] @ [64,32] + bf2 -> d_out
    for (int nt = 0; nt < 2; ++nt) {
        v8f c = {};
        for (int kt = 0; kt < 2; ++kt) {
            v16h a = pack_a_fast(&y1[wslot][0], 64, 0, kt * 32, lane);
            v16h b = pack_b16(Wf2S, 2, kt, nt, lane);
            c = __builtin_amdgcn_wmma_f32_16x16x32_f16(false, a, false, b, (short)0, c, false, false);
        }
        int col = nt * 16 + (lane & 15);
#pragma unroll
        for (int r = 0; r < 8; ++r) {
            int mrow = row0 + r + 8 * hi;
            if (live && mrow < n) out[(size_t)mrow * 32 + col] = c[r] + bf2[col];
        }
    }
}

// ---------------------------------------------------------------------------
extern "C" void kernel_launch(void* const* d_in, const int* in_sizes, int n_in,
                              void* d_out, int out_size, void* d_ws, size_t ws_size,
                              hipStream_t stream) {
    const float* uav = (const float*)d_in[0];
    const float* tf  = (const float*)d_in[1];
    const int*   ei  = (const int*)d_in[2];
    const float* adj = (const float*)d_in[3];
    const float* W1  = (const float*)d_in[4];
    const float* as1 = (const float*)d_in[5];
    const float* ad1 = (const float*)d_in[6];
    /* d_in[7] = b1 : cancels under training-mode BN (constant column shift) */
    const float* g1  = (const float*)d_in[8];
    const float* be1 = (const float*)d_in[9];
    const float* W2  = (const float*)d_in[10];
    const float* as2 = (const float*)d_in[11];
    const float* ad2 = (const float*)d_in[12];
    /* d_in[13] = b2 : cancels under BN */
    const float* g2  = (const float*)d_in[14];
    const float* be2 = (const float*)d_in[15];
    const float* Wt  = (const float*)d_in[16];
    /* d_in[17] = bt : cancels under BN */
    const float* tg  = (const float*)d_in[18];
    const float* tb  = (const float*)d_in[19];
    const float* Wf1 = (const float*)d_in[20];
    const float* bf1 = (const float*)d_in[21];
    const float* Wf2 = (const float*)d_in[22];
    const float* bf2 = (const float*)d_in[23];
    float* out = (float*)d_out;

    const int n = in_sizes[0] / 32;   // N UAVs
    const int t = in_sizes[1] / 16;   // T targets
    const long long E = in_sizes[2] / 2;
    const int* srcI = ei;
    const int* dstI = ei + E;

    // ---- workspace carve-up ----
    float* ws = (float*)d_ws;
    size_t o = 0;
    float*    h1    = ws + o; o += (size_t)n * 256;
    float*    x1    = ws + o; o += (size_t)n * 256;
    float*    aS1   = ws + o; o += (size_t)n * HEADS;
    float*    aD1   = ws + o; o += (size_t)n * HEADS;
    unsigned* m1    = (unsigned*)(ws + o); o += (size_t)n * HEADS;
    float*    den1  = ws + o; o += (size_t)n * HEADS;
    float*    ex1   = ws + o; o += (size_t)E * HEADS;
    float*    h2    = ws + o; o += (size_t)n * HDIM;
    float*    x2    = ws + o; o += (size_t)n * HDIM;
    float*    aS2   = ws + o; o += (size_t)n;
    float*    aD2   = ws + o; o += (size_t)n;
    unsigned* m2    = (unsigned*)(ws + o); o += (size_t)n;
    float*    den2  = ws + o; o += (size_t)n;
    float*    ex2   = ws + o; o += (size_t)E;
    float*    th    = ws + o; o += (size_t)t * HDIM;
    float*    comb  = ws + o; o += (size_t)n * 128;
    float*    tfeat = ws + o; o += (size_t)n * HDIM;
    float*    cntB  = ws + o; o += (size_t)n;
    float*    mu1   = ws + o; o += 256;
    float*    va1   = ws + o; o += 256;
    float*    mu2   = ws + o; o += 64;
    float*    va2   = ws + o; o += 64;
    float*    muT   = ws + o; o += 64;
    float*    vaT   = ws + o; o += 64;
    o = (o + 15) & ~(size_t)15;                  // 64B-align the f16 region
    auto halves_for = [](int K, int N) {
        return (size_t)((K + 31) / 32) * ((N + 15) / 16) * 32 * 16;
    };
    _Float16* f16base = (_Float16*)(ws + o);
    size_t ho = 0;
    _Float16* W1s  = f16base + ho; ho += halves_for(32, 256);
    _Float16* W2s  = f16base + ho; ho += halves_for(256, HDIM);
    _Float16* Wts  = f16base + ho; ho += halves_for(16, HDIM);
    _Float16* Wf1s = f16base + ho; ho += halves_for(128, HDIM);
    _Float16* Wf2s = f16base + ho; ho += halves_for(HDIM, 32);
    _Float16* thS  = f16base + ho; ho += halves_for(t, HDIM);

    auto cdivll = [](long long a, long long b) { return (unsigned)((a + b - 1) / b); };
    const bool nAligned = (n % 16 == 0);
    const bool tAligned = (t % 32 == 0);

    // ---- init accumulators ----
    fill_u32_kernel<<<cdivll((long long)n * 256, 256), 256, 0, stream>>>((unsigned*)x1, 0u, (long long)n * 256);
    fill_u32_kernel<<<cdivll((long long)n * HEADS, 256), 256, 0, stream>>>((unsigned*)den1, 0u, (long long)n * HEADS);
    fill_u32_kernel<<<cdivll((long long)n * HEADS, 256), 256, 0, stream>>>(m1, ENC_NEG_INF, (long long)n * HEADS);
    fill_u32_kernel<<<cdivll((long long)n * HDIM, 256), 256, 0, stream>>>((unsigned*)x2, 0u, (long long)n * HDIM);
    fill_u32_kernel<<<cdivll(n, 256), 256, 0, stream>>>((unsigned*)den2, 0u, (long long)n);
    fill_u32_kernel<<<cdivll(n, 256), 256, 0, stream>>>(m2, ENC_NEG_INF, (long long)n);
    fill_u32_kernel<<<cdivll((long long)n * HDIM, 256), 256, 0, stream>>>((unsigned*)tfeat, 0u, (long long)n * HDIM);
    fill_u32_kernel<<<cdivll(n, 256), 256, 0, stream>>>((unsigned*)cntB, 0u, (long long)n);

    // ---- pre-swizzle all weight B-operands to f16 WMMA layout ----
    b_swizzle_kernel<false><<<cdivll(1 * 16 * 32, 256), 256, 0, stream>>>(W1,  W1s,  32,  256);
    b_swizzle_kernel<false><<<cdivll(8 *  4 * 32, 256), 256, 0, stream>>>(W2,  W2s,  256, HDIM);
    b_swizzle_kernel<true ><<<cdivll(1 *  4 * 32, 256), 256, 0, stream>>>(Wt,  Wts,  16,  HDIM);
    b_swizzle_kernel<false><<<cdivll(4 *  4 * 32, 256), 256, 0, stream>>>(Wf1, Wf1s, 128, HDIM);
    b_swizzle_kernel<false><<<cdivll(2 *  2 * 32, 256), 256, 0, stream>>>(Wf2, Wf2s, HDIM, 32);

    // ---- GAT layer 1: h1 = uav @ W1  (M=n, N=256, K=32) ----
    if (nAligned) {
        long long waves = (long long)(n / 16) * (256 / 64);
        wmma_gemm_fast_kernel<<<cdivll(waves * 32, 256), 256, 0, stream>>>(uav, W1s, h1, nullptr, n, 256, 32);
    } else {
        long long waves = (long long)((n + 15) / 16) * 16;
        wmma_gemm_safe_kernel<<<cdivll(waves * 32, 256), 256, 0, stream>>>(uav, W1s, h1, nullptr, n, 256, 32);
    }
    att_kernel<<<cdivll((long long)n * HEADS, 256), 256, 0, stream>>>(h1, as1, ad1, aS1, aD1, n, HEADS);
    edge_max_kernel<<<cdivll(E * HEADS, 256), 256, 0, stream>>>(srcI, dstI, aS1, aD1, ex1, m1, E, HEADS);
    edge_exp_kernel<<<cdivll(E * HEADS, 256), 256, 0, stream>>>(dstI, ex1, m1, den1, E, HEADS);
    edge_msg_kernel<<<cdivll(E * HEADS * HDIM, 256), 256, 0, stream>>>(srcI, dstI, ex1, den1, h1, x1, E, HEADS);
    bn_stats_kernel<<<256, 256, 0, stream>>>(x1, mu1, va1, n, 256);
    bn_apply_kernel<<<cdivll((long long)n * 256, 256), 256, 0, stream>>>(x1, x1, mu1, va1, g1, be1, n, 256, 256, 0);

    // ---- GAT layer 2: h2 = x1 @ W2  (M=n, N=64, K=256) ----
    if (nAligned) {
        long long waves = (long long)(n / 16);
        wmma_gemm_fast_kernel<<<cdivll(waves * 32, 256), 256, 0, stream>>>(x1, W2s, h2, nullptr, n, HDIM, 256);
    } else {
        long long waves = (long long)((n + 15) / 16) * 4;
        wmma_gemm_safe_kernel<<<cdivll(waves * 32, 256), 256, 0, stream>>>(x1, W2s, h2, nullptr, n, HDIM, 256);
    }
    att_kernel<<<cdivll(n, 256), 256, 0, stream>>>(h2, as2, ad2, aS2, aD2, n, 1);
    edge_max_kernel<<<cdivll(E, 256), 256, 0, stream>>>(srcI, dstI, aS2, aD2, ex2, m2, E, 1);
    edge_exp_kernel<<<cdivll(E, 256), 256, 0, stream>>>(dstI, ex2, m2, den2, E, 1);
    edge_msg_kernel<<<cdivll(E * HDIM, 256), 256, 0, stream>>>(srcI, dstI, ex2, den2, h2, x2, E, 1);
    bn_stats_kernel<<<64, 256, 0, stream>>>(x2, mu2, va2, n, HDIM);
    // uav_h -> combined[:, 0:64]
    bn_apply_kernel<<<cdivll((long long)n * HDIM, 256), 256, 0, stream>>>(x2, comb, mu2, va2, g2, be2, n, HDIM, 128, 0);

    // ---- target branch: th = tf @ Wt  (M=t, N=64, K=16: swizzle pads K) ----
    {
        long long waves = (long long)((t + 15) / 16) * 4;
        wmma_gemm_safe_kernel<<<cdivll(waves * 32, 256), 256, 0, stream>>>(tf, Wts, th, nullptr, t, HDIM, 16);
    }
    bn_stats_kernel<<<64, 256, 0, stream>>>(th, muT, vaT, t, HDIM);
    bn_apply_kernel<<<cdivll((long long)t * HDIM, 256), 256, 0, stream>>>(th, th, muT, vaT, tg, tb, t, HDIM, HDIM, 1);
    // swizzle th (post-BN/relu) to f16 B layout for the pooling GEMM
    if (tAligned)
        b_swizzle_kernel<false><<<cdivll((long long)(t / 32) * 4 * 32, 256), 256, 0, stream>>>(th, thS, t, HDIM);
    else
        b_swizzle_kernel<true ><<<cdivll((long long)((t + 31) / 32) * 4 * 32, 256), 256, 0, stream>>>(th, thS, t, HDIM);

    // ---- fused masked mean pooling (K-split) -> tfeat/cnt, then normalize ----
    {
        long long waves = (long long)((n + 15) / 16) * KSPLIT;
        if (nAligned && tAligned)
            pool_kernel<false><<<cdivll(waves * 32, 256), 256, 0, stream>>>(adj, thS, tfeat, cntB, n, t);
        else
            pool_kernel<true><<<cdivll(waves * 32, 256), 256, 0, stream>>>(adj, thS, tfeat, cntB, n, t);
        pool_norm_kernel<<<cdivll((long long)n * HDIM, 256), 256, 0, stream>>>(tfeat, cntB, comb, n);
    }

    // ---- fused output MLP -> d_out ----
    {
        long long waves = (long long)((n + 15) / 16);
        if (nAligned)
            mlp_kernel<false><<<cdivll(waves * 32, 256), 256, 0, stream>>>(comb, Wf1s, bf1, Wf2s, bf2, out, n);
        else
            mlp_kernel<true><<<cdivll(waves * 32, 256), 256, 0, stream>>>(comb, Wf1s, bf1, Wf2s, bf2, out, n);
    }
}
